// GCN_G_70961449664987
// MI455X (gfx1250) — matrix operand order repaired and checked by reading, not compiled, not run on hardware
//
#include <hip/hip_runtime.h>
#include <hip/hip_fp16.h>
#include <math.h>

typedef __attribute__((ext_vector_type(16))) _Float16 v16h;
typedef __attribute__((ext_vector_type(8)))  float    v8f;

#define NNODE 100000
#define NEDGE 1600000
#define FIN   64
#define HID   128
#define NG    64
#define NTILES (HID / 16)   // 8 output column tiles

__device__ __forceinline__ void atomAddF(float* p, float v) {
  __hip_atomic_fetch_add(p, v, __ATOMIC_RELAXED, __HIP_MEMORY_SCOPE_AGENT);
}

// ---------------- degree / normalization ----------------
__global__ void k_fill(float* p, float v, int n) {
  int t = blockIdx.x * blockDim.x + threadIdx.x;
  if (t < n) p[t] = v;
}

__global__ void k_deg_atomic(const int* __restrict__ dst, float* __restrict__ deg, int e) {
  int t = blockIdx.x * blockDim.x + threadIdx.x;
  if (t < e) atomAddF(deg + dst[t], 1.0f);
}

__global__ void k_rsqrt_inplace(float* p, int n) {
  int t = blockIdx.x * blockDim.x + threadIdx.x;
  if (t < n) p[t] = rsqrtf(fmaxf(p[t], 1.0f));
}

// ---------------- WMMA fragment packing ----------------
// B fragment (16-bit B 32x16, ISA 7.12.2): lane L holds column N=L&15,
// kb=(L<16?0:8); i<8 -> K=ktile*32+kb+i ; i>=8 -> K=ktile*32+16+kb+(i-8).
__global__ void k_packB(const float* __restrict__ W, _Float16* __restrict__ out,
                        int Ktiles, int Ntiles, int Nout, int total) {
  int t = blockIdx.x * blockDim.x + threadIdx.x;
  if (t >= total) return;
  int i    = t & 15;
  int lane = (t >> 4) & 31;
  int tile = t >> 9;                 // kt*Ntiles + nt
  int nt = tile % Ntiles;
  int kt = tile / Ntiles;
  int k = kt * 32 + ((i < 8) ? (((lane & 16) ? 8 : 0) + i) : (16 + ((lane & 16) ? 8 : 0) + (i - 8)));
  int n = nt * 16 + (lane & 15);
  out[t] = (_Float16)W[(size_t)k * Nout + n];
}

// A fragment (16-bit A 16x32, ISA 7.12.2): lane L holds row M=L&15,
// kb=(L<16?0:8); i<8 -> K=kb+i ; i>=8 -> K=16+kb+(i-8).
__global__ void k_packA(const float* __restrict__ X, _Float16* __restrict__ out,
                        int Ktiles, int K, int total) {
  int t = blockIdx.x * blockDim.x + threadIdx.x;
  if (t >= total) return;
  int i    = t & 15;
  int lane = (t >> 4) & 31;
  int tile = t >> 9;                 // mt*Ktiles + kt
  int kt = tile % Ktiles;
  int mt = tile / Ktiles;
  int kb = (lane & 16) ? 8 : 0;
  int kk = (i < 8) ? (kb + i) : (16 + kb + (i - 8));
  int m = mt * 16 + (lane & 15);
  out[t] = (_Float16)X[(size_t)m * K + (kt * 32 + kk)];
}

// ---------------- WMMA GEMM: C[M, HID] = A * B ----------------
// One wave per 16-row M strip: A fragments loaded ONCE into registers,
// all 8 N-tiles computed from them (B is tiny and L2-resident).
template <int KT>
__global__ void k_gemm(const _Float16* __restrict__ Ap, const _Float16* __restrict__ Bp,
                       float* __restrict__ C, int Mtiles) {
  int mt   = (blockIdx.x * blockDim.x + threadIdx.x) >> 5;   // wave-uniform
  int lane = threadIdx.x & 31;
  if (mt >= Mtiles) return;                                  // uniform exit: EXEC stays all-1
  const v16h* Av = (const v16h*)Ap;
  const v16h* Bv = (const v16h*)Bp;

  v16h a[KT];
#pragma unroll
  for (int kt = 0; kt < KT; ++kt)
    a[kt] = Av[((size_t)(mt * KT + kt)) * 32 + lane];

  const int Nout = NTILES * 16;                              // 128
  int row0 = mt * 16 + ((lane & 16) ? 8 : 0);
  int col0 = lane & 15;

#pragma unroll
  for (int nt = 0; nt < NTILES; ++nt) {
    v8f acc = {};
#pragma unroll
    for (int kt = 0; kt < KT; ++kt) {
      v16h b = Bv[((size_t)(kt * NTILES + nt)) * 32 + lane];
      acc = __builtin_amdgcn_wmma_f32_16x16x32_f16(false, a[kt], false, b, (short)0, acc,
                                                   false, false);
    }
    float* o = C + (size_t)row0 * Nout + nt * 16 + col0;
#pragma unroll
    for (int r = 0; r < 8; ++r) o[(size_t)r * Nout] = acc[r];
  }
}

// ---------------- bias init / scatter / elu ----------------
__global__ void k_set_bias(float* __restrict__ H, const float* __restrict__ b, int total) {
  int t = blockIdx.x * blockDim.x + threadIdx.x;
  if (t < total) H[t] = b[t & (HID - 1)];
}

// One wave per (edge or self-loop); lane l covers features 4l..4l+3.
__global__ void k_scatter(const float* __restrict__ T, float* __restrict__ H,
                          const int* __restrict__ srcA, const int* __restrict__ dstA,
                          const float* __restrict__ dinv, int e, int n) {
  int g = blockIdx.x * blockDim.x + threadIdx.x;
  int edge = g >> 5;
  int lane = g & 31;
  if (edge >= e + n) return;
  int s, d;
  if (edge < e) { s = srcA[edge]; d = dstA[edge]; }
  else          { s = d = edge - e; }
  float nrm = dinv[s] * dinv[d];
  float4 hv = ((const float4*)(T + (size_t)s * HID))[lane];
  float* o = H + (size_t)d * HID + lane * 4;
  atomAddF(o + 0, hv.x * nrm);
  atomAddF(o + 1, hv.y * nrm);
  atomAddF(o + 2, hv.z * nrm);
  atomAddF(o + 3, hv.w * nrm);
}

__global__ void k_elu(float* __restrict__ H, int total) {
  int t = blockIdx.x * blockDim.x + threadIdx.x;
  if (t < total) {
    float x = H[t];
    H[t] = (x > 0.0f) ? x : expm1f(x);
  }
}

// ---------------- pooling ----------------
__global__ void k_pool(const float* __restrict__ H, const int* __restrict__ batch,
                       float* __restrict__ pooled, float* __restrict__ cnt, int n) {
  int g = blockIdx.x * blockDim.x + threadIdx.x;
  int node = g >> 5;
  int lane = g & 31;
  if (node >= n) return;
  int grp = batch[node];
  float4 hv = ((const float4*)(H + (size_t)node * HID))[lane];
  float* o = pooled + (size_t)grp * HID + lane * 4;
  atomAddF(o + 0, hv.x);
  atomAddF(o + 1, hv.y);
  atomAddF(o + 2, hv.z);
  atomAddF(o + 3, hv.w);
  if (lane == 0) atomAddF(cnt + grp, 1.0f);
}

// ---------------- MLP head + log_softmax (tiny) ----------------
__global__ void k_head(const float* __restrict__ pooled, const float* __restrict__ cnt,
                       const float* __restrict__ stats,
                       const float* __restrict__ Wf1, const float* __restrict__ bf1,
                       const float* __restrict__ Wf2, const float* __restrict__ bf2,
                       float* __restrict__ out) {
  int g = threadIdx.x;
  if (g >= NG) return;
  float cinv = 1.0f / fmaxf(cnt[g], 1.0f);
  float z1[20];
#pragma unroll 4
  for (int j = 0; j < 20; ++j) {
    float s = bf1[j];
    for (int i = 0; i < HID; ++i) s += pooled[g * HID + i] * cinv * Wf1[i * 20 + j];
    for (int i = 0; i < 10;  ++i) s += stats[g * 10 + i] * Wf1[(HID + i) * 20 + j];
    z1[j] = fmaxf(s, 0.0f);
  }
  float z2[5];
  float m = -1e30f;
#pragma unroll
  for (int k = 0; k < 5; ++k) {
    float s = bf2[k];
    for (int j = 0; j < 20; ++j) s += z1[j] * Wf2[j * 5 + k];
    z2[k] = s;
    m = fmaxf(m, s);
  }
  float se = 0.0f;
#pragma unroll
  for (int k = 0; k < 5; ++k) se += expf(z2[k] - m);
  float lse = m + logf(se);
#pragma unroll
  for (int k = 0; k < 5; ++k) out[g * 5 + k] = z2[k] - lse;
}

// ---------------- orchestration ----------------
extern "C" void kernel_launch(void* const* d_in, const int* in_sizes, int n_in,
                              void* d_out, int out_size, void* d_ws, size_t ws_size,
                              hipStream_t stream) {
  const float* x     = (const float*)d_in[0];
  const int*   eidx  = (const int*)d_in[1];   // [2, E] flat
  const int*   batch = (const int*)d_in[2];
  /* d_in[3] = eig, unused by reference */
  const float* stats = (const float*)d_in[4];
  const float* W1  = (const float*)d_in[5];
  const float* b1  = (const float*)d_in[6];
  const float* W2  = (const float*)d_in[7];
  const float* b2  = (const float*)d_in[8];
  const float* Wf1 = (const float*)d_in[9];
  const float* bf1 = (const float*)d_in[10];
  const float* Wf2 = (const float*)d_in[11];
  const float* bf2 = (const float*)d_in[12];
  (void)in_sizes; (void)n_in; (void)out_size; (void)ws_size;

  const int* srcA = eidx;
  const int* dstA = eidx + NEDGE;

  // workspace carve-up (~129 MB)
  char* ws = (char*)d_ws;
  size_t off = 0;
  auto take = [&](size_t bytes) -> char* {
    char* p = ws + off;
    off += (bytes + 255) & ~(size_t)255;
    return p;
  };
  float*    dinv   = (float*)take((size_t)NNODE * 4);
  _Float16* W1p    = (_Float16*)take((size_t)FIN * HID * 2);
  _Float16* W2p    = (_Float16*)take((size_t)HID * HID * 2);
  _Float16* Apack  = (_Float16*)take((size_t)NNODE * HID * 2);
  float*    bufT   = (float*)take((size_t)NNODE * HID * 4);
  float*    bufH   = (float*)take((size_t)NNODE * HID * 4);
  float*    pooled = (float*)take((size_t)NG * HID * 4);
  float*    cnt    = (float*)take((size_t)NG * 4);

  const int Mtiles = NNODE / 16;   // 6250
  const int TB = 256;
  auto blks = [](long long n, int tb) { return (unsigned)((n + tb - 1) / tb); };

  // 1. degree (incl. self loop) -> dinv
  k_fill<<<blks(NNODE, TB), TB, 0, stream>>>(dinv, 1.0f, NNODE);
  k_deg_atomic<<<blks(NEDGE, TB), TB, 0, stream>>>(dstA, dinv, NEDGE);
  k_rsqrt_inplace<<<blks(NNODE, TB), TB, 0, stream>>>(dinv, NNODE);

  // 2. pack weights + x into WMMA fragment layouts
  {
    int totB1 = 2 * NTILES * 512;
    k_packB<<<blks(totB1, TB), TB, 0, stream>>>(W1, W1p, 2, NTILES, HID, totB1);
    int totB2 = 4 * NTILES * 512;
    k_packB<<<blks(totB2, TB), TB, 0, stream>>>(W2, W2p, 4, NTILES, HID, totB2);
    long long totA1 = (long long)Mtiles * 2 * 512;
    k_packA<<<blks(totA1, TB), TB, 0, stream>>>(x, Apack, 2, FIN, (int)totA1);
  }

  // 3. GEMM1: bufT = x @ W1   (one wave per 16-row strip, 16 v_wmma each)
  k_gemm<2><<<blks((long long)Mtiles * 32, TB), TB, 0, stream>>>(Apack, W1p, bufT, Mtiles);

  // 4. conv1 aggregate: bufH = b1 + scatter(norm * bufT[src] -> dst), then ELU
  k_set_bias<<<blks((long long)NNODE * HID, TB), TB, 0, stream>>>(bufH, b1, NNODE * HID);
  k_scatter<<<blks((long long)(NEDGE + NNODE) * 32, TB), TB, 0, stream>>>(
      bufT, bufH, srcA, dstA, dinv, NEDGE, NNODE);
  k_elu<<<blks((long long)NNODE * HID, TB), TB, 0, stream>>>(bufH, NNODE * HID);

  // 5. GEMM2: bufT = bufH @ W2
  {
    long long totA2 = (long long)Mtiles * 4 * 512;
    k_packA<<<blks(totA2, TB), TB, 0, stream>>>(bufH, Apack, 4, HID, (int)totA2);
  }
  k_gemm<4><<<blks((long long)Mtiles * 32, TB), TB, 0, stream>>>(Apack, W2p, bufT, Mtiles);

  // 6. conv2 aggregate: bufH = b2 + scatter(...)
  k_set_bias<<<blks((long long)NNODE * HID, TB), TB, 0, stream>>>(bufH, b2, NNODE * HID);
  k_scatter<<<blks((long long)(NEDGE + NNODE) * 32, TB), TB, 0, stream>>>(
      bufT, bufH, srcA, dstA, dinv, NEDGE, NNODE);

  // 7. mean pool per graph
  k_fill<<<blks(NG * HID, TB), TB, 0, stream>>>(pooled, 0.0f, NG * HID);
  k_fill<<<1, NG, 0, stream>>>(cnt, 0.0f, NG);
  k_pool<<<blks((long long)NNODE * 32, TB), TB, 0, stream>>>(bufH, batch, pooled, cnt, NNODE);

  // 8. head MLP + log_softmax -> d_out [64,5]
  k_head<<<1, NG, 0, stream>>>(pooled, cnt, stats, Wf1, bf1, Wf2, bf2, (float*)d_out);
}
